// Int8MixtralSparseMoeBlock_75368086110386
// MI455X (gfx1250) — 8-bit matrix operand order repaired and checked by reading, compiled, not run on hardware
//
#include <hip/hip_runtime.h>
#include <hip/hip_bf16.h>

#define T_TOK 8192
#define HDIM  1024
#define FDIM  3584
#define EDIM  8
#define MT    64

#define XQ_OFF   0
#define GQ_OFF   65536
#define XS_OFF   (65536 + 229376)
#define WT_OFF   (XS_OFF + 256)
#define TK_OFF   (WT_OFF + 256)
#define STG_OFF  (TK_OFF + 256)
#define SMEM_TOTAL (STG_OFF + 8 * 2048)

typedef int v8i __attribute__((ext_vector_type(8)));
typedef unsigned int u32x4 __attribute__((ext_vector_type(4)));
typedef int i32x4 __attribute__((ext_vector_type(4)));
typedef int i32x8 __attribute__((ext_vector_type(8)));

#define HAVE_TDM __has_builtin(__builtin_amdgcn_tensor_load_to_lds)

__device__ __forceinline__ v8i v8i_zero() {
  v8i z;
#pragma unroll
  for (int i = 0; i < 8; ++i) z[i] = 0;
  return z;
}

__device__ __forceinline__ int clampi(int v, int lo, int hi) {
  return v < lo ? lo : (v > hi ? hi : v);
}

__device__ __forceinline__ v8i wmma_i8(v8i a, v8i b, v8i c) {
  return __builtin_amdgcn_wmma_i32_16x16x64_iu8(true, a, true, b, c, false, false);
}

__device__ __forceinline__ void async_cp_b128(unsigned lds_off, unsigned long long gaddr) {
  asm volatile("global_load_async_to_lds_b128 %0, %1, off"
               :: "v"(lds_off), "v"(gaddr) : "memory");
}

__device__ __forceinline__ void wait_async0() {
#if __has_builtin(__builtin_amdgcn_s_wait_asynccnt)
  __builtin_amdgcn_s_wait_asynccnt(0);
#else
  asm volatile("s_wait_asynccnt 0x0" ::: "memory");
#endif
}

__device__ __forceinline__ void wait_async2() {
  asm volatile("s_wait_asynccnt 0x2" ::: "memory");
}

__device__ __forceinline__ void wait_ds0() {
  asm volatile("s_wait_dscnt 0x0" ::: "memory");
}

__device__ __forceinline__ void stage_tile_async(const signed char* wrow, int K,
                                                 unsigned lds_off, int lane) {
#pragma unroll
  for (int j = 0; j < 2; ++j) {
    int u = j * 32 + lane;
    unsigned lo = lds_off + (unsigned)(u << 4);
    const signed char* g = wrow + (long long)(u >> 2) * K + ((u & 3) << 4);
    async_cp_b128(lo, (unsigned long long)(uintptr_t)g);
  }
}

__device__ __forceinline__ v8i lda_lds(const signed char* tile, int m0, int k0,
                                       int stride, int lane) {
  int m  = m0 + (lane & 15);
  int kb = k0 + ((lane >> 4) << 4);
  const signed char* row = tile + m * stride + kb;
  v8i a;
#pragma unroll
  for (int i = 0; i < 8; ++i) {
    int off = ((i & 3) << 2) + ((i >> 2) << 5);
    a[i] = *(const int*)(row + off);
  }
  return a;
}

__device__ __forceinline__ v8i ldb_lds(const signed char* tile, int lane) {
  const signed char* row = tile + ((lane & 15) << 6) + ((lane >> 4) << 4);
  int4 lo = *(const int4*)(row);
  int4 hi = *(const int4*)(row + 32);
  v8i b;
  b[0] = lo.x; b[1] = lo.y; b[2] = lo.z; b[3] = lo.w;
  b[4] = hi.x; b[5] = hi.y; b[6] = hi.z; b[7] = hi.w;
  return b;
}

__global__ __launch_bounds__(256)
void quantize_w_kernel(const float* __restrict__ src, unsigned int* __restrict__ dst,
                       long long n4) {
  long long i = (long long)blockIdx.x * 256 + threadIdx.x;
  if (i >= n4) return;
  float4 v = ((const float4*)src)[i];
  int a = clampi((int)rintf(v.x), -127, 127);
  int b = clampi((int)rintf(v.y), -127, 127);
  int c = clampi((int)rintf(v.z), -127, 127);
  int d = clampi((int)rintf(v.w), -127, 127);
  dst[i] = (unsigned int)(a & 255) | ((unsigned int)(b & 255) << 8) |
           ((unsigned int)(c & 255) << 16) | ((unsigned int)(d & 255) << 24);
}

__global__ __launch_bounds__(256)
void router_quant_kernel(const float* __restrict__ x, const float* __restrict__ gw,
                         signed char* __restrict__ xq, float* __restrict__ xs,
                         int* __restrict__ cnt, int* __restrict__ lists,
                         float* __restrict__ wts, float* __restrict__ logits) {
  int tok  = blockIdx.x * 8 + (threadIdx.x >> 5);
  int lane = threadIdx.x & 31;
  const float* xr = x + (long long)tok * HDIM;

  float acc[EDIM];
#pragma unroll
  for (int e = 0; e < EDIM; ++e) acc[e] = 0.f;
  float amax = 0.f;

  for (int h = lane; h < HDIM; h += 32) {
    float v = xr[h];
    amax = fmaxf(amax, fabsf(v));
#pragma unroll
    for (int e = 0; e < EDIM; ++e) acc[e] += v * gw[e * HDIM + h];
  }
#pragma unroll
  for (int off = 16; off > 0; off >>= 1) {
    amax = fmaxf(amax, __shfl_xor(amax, off, 32));
#pragma unroll
    for (int e = 0; e < EDIM; ++e) acc[e] += __shfl_xor(acc[e], off, 32);
  }

  float s = fmaxf(amax, 1e-8f) / 127.f;
  float inv_s = 1.f / s;
  for (int h = lane; h < HDIM; h += 32) {
    int q = clampi((int)rintf(xr[h] * inv_s), -127, 127);
    xq[(long long)tok * HDIM + h] = (signed char)q;
  }

  if (lane == 0) {
    xs[tok] = s;
#pragma unroll
    for (int e = 0; e < EDIM; ++e) logits[tok * EDIM + e] = acc[e];
    float mx = acc[0];
#pragma unroll
    for (int e = 1; e < EDIM; ++e) mx = fmaxf(mx, acc[e]);
    float ex[EDIM], sum = 0.f;
#pragma unroll
    for (int e = 0; e < EDIM; ++e) { ex[e] = __expf(acc[e] - mx); sum += ex[e]; }
    int i0 = 0;
#pragma unroll
    for (int e = 1; e < EDIM; ++e) if (acc[e] > acc[i0]) i0 = e;
    int i1 = (i0 == 0) ? 1 : 0;
#pragma unroll
    for (int e = 0; e < EDIM; ++e)
      if (e != i0 && acc[e] > acc[i1]) i1 = e;
    float p0 = ex[i0] / sum, p1 = ex[i1] / sum;
    float rn = 1.f / (p0 + p1);
    int s0 = atomicAdd(&cnt[i0], 1);
    lists[i0 * T_TOK + s0] = tok;  wts[i0 * T_TOK + s0] = p0 * rn;
    int s1 = atomicAdd(&cnt[i1], 1);
    lists[i1 * T_TOK + s1] = tok;  wts[i1 * T_TOK + s1] = p1 * rn;
  }
}

__global__ __launch_bounds__(256)
void moe_expert_kernel(const signed char* __restrict__ xq, const float* __restrict__ xs,
                       const signed char* __restrict__ w1, const float* __restrict__ w1s,
                       const signed char* __restrict__ w3, const float* __restrict__ w3s,
                       const signed char* __restrict__ w2, const float* __restrict__ w2s,
                       const float* __restrict__ dscale,
                       const int* __restrict__ cnt, const int* __restrict__ lists,
                       const float* __restrict__ wts, float* __restrict__ out) {
  extern __shared__ char smem[];
  signed char* xq_s = (signed char*)(smem + XQ_OFF);
  signed char* gq_s = (signed char*)(smem + GQ_OFF);
  float* xs_s  = (float*)(smem + XS_OFF);
  float* wt_s  = (float*)(smem + WT_OFF);
  int*   tok_s = (int*)(smem + TK_OFF);

  int e     = blockIdx.y;
  int n_e   = cnt[e];
  int tile0 = blockIdx.x * MT;
  if (tile0 >= n_e) return;

  int tid  = threadIdx.x;
  int wave = tid >> 5, lane = tid & 31;

  if (tid < MT) {
    int r  = tile0 + tid;
    int tk = 0; float w = 0.f;
    if (r < n_e) { tk = lists[e * T_TOK + r]; w = wts[e * T_TOK + r]; }
    tok_s[tid] = tk; wt_s[tid] = w; xs_s[tid] = xs[tk];
  }

#if HAVE_TDM
  if (wave < 4) {
    unsigned short idx[16];
#pragma unroll
    for (int j = 0; j < 16; ++j) {
      int r = tile0 + wave * 16 + j;
      idx[j] = (unsigned short)((r < n_e) ? lists[e * T_TOK + r] : 0);
    }
    unsigned long long ga = (unsigned long long)(uintptr_t)xq;
    u32x4 g0;
    g0[0] = 1u | (0u << 30) | (1u << 31);
    g0[1] = (unsigned)(XQ_OFF + wave * 16 * HDIM);
    g0[2] = (unsigned)(ga & 0xffffffffu);
    g0[3] = (unsigned)((ga >> 32) & 0x1ffffffu) | (2u << 30);
    i32x8 g1;
    g1[0] = 0;
    g1[1] = (int)(((unsigned)HDIM & 0xffffu) << 16);
    g1[2] = (int)((((unsigned)HDIM >> 16) & 0xffffu) |
                  (((unsigned)T_TOK & 0xffffu) << 16));
    g1[3] = (int)(((unsigned)T_TOK >> 16) |
                  (((unsigned)HDIM & 0xffffu) << 16));
    g1[4] = 16;
    g1[5] = HDIM;
    g1[6] = 0; g1[7] = 0;
    i32x4 g2, g3;
#pragma unroll
    for (int j = 0; j < 4; ++j) {
      g2[j] = (int)((unsigned)idx[2 * j] | ((unsigned)idx[2 * j + 1] << 16));
      g3[j] = (int)((unsigned)idx[8 + 2 * j] | ((unsigned)idx[9 + 2 * j] << 16));
    }
    i32x8 g4 = {0, 0, 0, 0, 0, 0, 0, 0};
    __builtin_amdgcn_tensor_load_to_lds(g0, g1, g2, g3, g4, 0);
#if __has_builtin(__builtin_amdgcn_s_wait_tensorcnt)
    __builtin_amdgcn_s_wait_tensorcnt(0);
#else
    asm volatile("s_wait_tensorcnt 0x0" ::: "memory");
#endif
  }
  __syncthreads();
#else
  __syncthreads();
  {
    int row = tid >> 2, part = tid & 3;
    int r = tile0 + row;
    uint4* dst = (uint4*)(xq_s + row * HDIM) + part * 16;
    if (r < n_e) {
      const uint4* src = (const uint4*)(xq + (long long)tok_s[row] * HDIM) + part * 16;
#pragma unroll
      for (int i = 0; i < 16; ++i) dst[i] = src[i];
    } else {
      uint4 z = {0u, 0u, 0u, 0u};
#pragma unroll
      for (int i = 0; i < 16; ++i) dst[i] = z;
    }
  }
  __syncthreads();
#endif

  signed char* stg   = (signed char*)(smem + STG_OFF) + wave * 2048;
  unsigned     stg_o = (unsigned)(STG_OFF + wave * 2048);
  float dsc = dscale[e];
  float inv_dsc = 1.0f / dsc;

  const signed char* w1e = w1 + (long long)e * FDIM * HDIM;
  const signed char* w3e = w3 + (long long)e * FDIM * HDIM;
  for (int nt = wave; nt < FDIM / 16; nt += 8) {
    const signed char* w1t = w1e + (long long)(nt * 16) * HDIM;
    const signed char* w3t = w3e + (long long)(nt * 16) * HDIM;

    v8i acc1[4], acc3[4];
#pragma unroll
    for (int m = 0; m < 4; ++m) { acc1[m] = v8i_zero(); acc3[m] = v8i_zero(); }

    wait_ds0();
    stage_tile_async(w1t, HDIM, stg_o, lane);
    stage_tile_async(w3t, HDIM, stg_o + 1024, lane);

    for (int kc = 0; kc < HDIM; kc += 64) {
      v8i a[4];
#pragma unroll
      for (int m = 0; m < 4; ++m) a[m] = lda_lds(xq_s, m * 16, kc, HDIM, lane);
      wait_async0();
      v8i b1 = ldb_lds(stg, lane);
      v8i b3 = ldb_lds(stg + 1024, lane);
      if (kc + 64 < HDIM) {
        wait_ds0();
        stage_tile_async(w1t + kc + 64, HDIM, stg_o, lane);
        stage_tile_async(w3t + kc + 64, HDIM, stg_o + 1024, lane);
      }
#pragma unroll
      for (int m = 0; m < 4; ++m) acc1[m] = wmma_i8(a[m], b1, acc1[m]);
#pragma unroll
      for (int m = 0; m < 4; ++m) acc3[m] = wmma_i8(a[m], b3, acc3[m]);
    }
    int fcol = nt * 16 + (lane & 15);
    float s1 = w1s[e * FDIM + fcol];
    float s3 = w3s[e * FDIM + fcol];
    int mhi = (lane >> 4) << 3;
#pragma unroll
    for (int m = 0; m < 4; ++m) {
#pragma unroll
      for (int r = 0; r < 8; ++r) {
        int mloc = m * 16 + mhi + r;
        float xsv = xs_s[mloc];
        float h1 = (float)acc1[m][r] * xsv * s1;
        float h3 = (float)acc3[m][r] * xsv * s3;
        float g = (h1 / (1.0f + __expf(-h1))) * h3;
        int q = clampi((int)rintf(g * inv_dsc), -127, 127);
        gq_s[mloc * FDIM + fcol] = (signed char)q;
      }
    }
  }
  __syncthreads();

  const signed char* w2e = w2 + (long long)e * HDIM * FDIM;
  signed char* stg2   = (signed char*)(smem + XQ_OFF) + wave * 2048;
  unsigned     stg2_o = (unsigned)(XQ_OFF + wave * 2048);
  for (int nt = wave; nt < HDIM / 16; nt += 8) {
    const signed char* w2t = w2e + (long long)(nt * 16) * FDIM;

    v8i acc[4];
#pragma unroll
    for (int m = 0; m < 4; ++m) acc[m] = v8i_zero();

    wait_ds0();
    stage_tile_async(w2t, FDIM, stg2_o, lane);
    int buf = 0;

    for (int kc = 0; kc < FDIM; kc += 64) {
      if (kc + 64 < FDIM) {
        wait_ds0();
        stage_tile_async(w2t + kc + 64, FDIM, stg2_o + ((unsigned)(buf ^ 1) << 10), lane);
        wait_async2();
      } else {
        wait_async0();
      }
      v8i b = ldb_lds(stg2 + (buf << 10), lane);
      v8i a[4];
#pragma unroll
      for (int m = 0; m < 4; ++m) a[m] = lda_lds(gq_s, m * 16, kc, FDIM, lane);
#pragma unroll
      for (int m = 0; m < 4; ++m) acc[m] = wmma_i8(a[m], b, acc[m]);
      buf ^= 1;
    }
    int hcol = nt * 16 + (lane & 15);
    float sc = dsc * w2s[e * HDIM + hcol];
    int mhi = (lane >> 4) << 3;
#pragma unroll
    for (int m = 0; m < 4; ++m) {
#pragma unroll
      for (int r = 0; r < 8; ++r) {
        int mloc = m * 16 + mhi + r;
        float y = (float)acc[m][r] * sc;
        atomicAdd(out + (long long)tok_s[mloc] * HDIM + hcol, wt_s[mloc] * y);
      }
    }
  }
}

extern "C" void kernel_launch(void* const* d_in, const int* in_sizes, int n_in,
                              void* d_out, int out_size, void* d_ws, size_t ws_size,
                              hipStream_t stream) {
  (void)in_sizes; (void)n_in; (void)out_size; (void)ws_size;
  const float* x    = (const float*)d_in[0];
  const float* gw   = (const float*)d_in[1];
  const float* w1f  = (const float*)d_in[2];
  const float* w1sc = (const float*)d_in[3];
  const float* w3f  = (const float*)d_in[4];
  const float* w3sc = (const float*)d_in[5];
  const float* w2f  = (const float*)d_in[6];
  const float* w2sc = (const float*)d_in[7];
  const float* dsc  = (const float*)d_in[8];
  float* out    = (float*)d_out;
  float* logits = out + (size_t)T_TOK * HDIM;

  char* ws = (char*)d_ws;
  size_t off = 0;
  auto carve = [&](size_t bytes) -> char* {
    char* p = ws + off;
    off = (off + bytes + 255) & ~(size_t)255;
    return p;
  };
  const size_t WBYTES = (size_t)EDIM * FDIM * HDIM;
  signed char* w1q = (signed char*)carve(WBYTES);
  signed char* w3q = (signed char*)carve(WBYTES);
  signed char* w2q = (signed char*)carve(WBYTES);
  signed char* xq  = (signed char*)carve((size_t)T_TOK * HDIM);
  float*       xs  = (float*)carve((size_t)T_TOK * sizeof(float));
  int*         cnt = (int*)carve(EDIM * sizeof(int));
  int*       lists = (int*)carve((size_t)EDIM * T_TOK * sizeof(int));
  float*       wts = (float*)carve((size_t)EDIM * T_TOK * sizeof(float));

  (void)hipMemsetAsync(out, 0, (size_t)T_TOK * HDIM * sizeof(float), stream);
  (void)hipMemsetAsync(cnt, 0, EDIM * sizeof(int), stream);

  long long n4 = (long long)EDIM * FDIM * HDIM / 4;
  int cvblocks = (int)((n4 + 255) / 256);
  quantize_w_kernel<<<cvblocks, 256, 0, stream>>>(w1f, (unsigned int*)w1q, n4);
  quantize_w_kernel<<<cvblocks, 256, 0, stream>>>(w3f, (unsigned int*)w3q, n4);
  quantize_w_kernel<<<cvblocks, 256, 0, stream>>>(w2f, (unsigned int*)w2q, n4);

  router_quant_kernel<<<T_TOK / 8, 256, 0, stream>>>(x, gw, xq, xs, cnt, lists, wts, logits);

  moe_expert_kernel<<<dim3(T_TOK / MT, EDIM), 256, SMEM_TOTAL, stream>>>(
      xq, xs, w1q, w1sc, w3q, w3sc, w2q, w2sc, dsc, cnt, lists, wts, out);
}
